// Attention_5111011082736
// MI455X (gfx1250) — matrix operand order repaired and checked by reading, not compiled, not run on hardware
//
#include <hip/hip_runtime.h>
#include <hip/hip_bf16.h>

// ---------------------------------------------------------------------------
// CDNA5 (gfx1250) attention block: LN+FiLM -> QKV GEMM -> flash attention with
// ALiBi + entropy loss -> out projection.
// GEMMs: v_wmma_f32_16x16x32_bf16.  Attention K/V tiles: TDM tensor_load_to_lds
// double-buffered, tracked with TENSORcnt; fragments served from LDS.
// ---------------------------------------------------------------------------

typedef __attribute__((ext_vector_type(16))) __bf16 v16bf;
typedef __attribute__((ext_vector_type(8)))  float  v8f;
typedef __attribute__((ext_vector_type(4)))  unsigned int v4u;
typedef __attribute__((ext_vector_type(8)))  int    v8i;
typedef __attribute__((ext_vector_type(4)))  int    v4i;

#define B_    4
#define N_    2048
#define D_    512
#define H_    4
#define DH_   128
#define ROWS_ (B_ * N_)          // 8192
#define NQKV_ (3 * D_)           // 1536

__device__ __forceinline__ unsigned short f32_to_bf16(float f) {
    unsigned int u = __float_as_uint(f);
    u = (u + 0x7FFFu + ((u >> 16) & 1u)) >> 16;   // round-to-nearest-even
    return (unsigned short)u;
}

__device__ __forceinline__ v8f wmma_bf16(v16bf a, v16bf b, v8f c) {
    return __builtin_amdgcn_wmma_f32_16x16x32_bf16(false, a, false, b, (short)0, c,
                                                   false, false);
}

// A-matrix fragment (16x32 bf16, ISA 7.12.2): lane row = lane&15,
// K per VGPR: grp0 -> {0..7,16..23}, grp1 -> {8..15,24..31}.
__device__ __forceinline__ v16bf load_a16(const unsigned short* rowptr, int k0, int grp) {
    union { unsigned int u[8]; v16bf v; } cv;
#pragma unroll
    for (int vv = 0; vv < 8; ++vv) {
        int kk = ((vv & 4) << 2) + grp * 8 + ((vv & 3) << 1);
        cv.u[vv] = *(const unsigned int*)(rowptr + k0 + kk);
    }
    return cv.v;
}

// B-matrix fragment (32x16 bf16): lane col = lane&15, K per element same as A.
__device__ __forceinline__ v16bf load_b16(const unsigned short* p, int stride, int grp) {
    union { unsigned short u[16]; v16bf v; } cv;
#pragma unroll
    for (int s = 0; s < 16; ++s) {
        int kk = ((s & 8) << 1) + grp * 8 + (s & 7);
        cv.u[s] = p[(long)kk * stride];
    }
    return cv.v;
}

// Low 32 bits of a generic LDS pointer == LDS byte offset (ISA 10.2 aperture rule).
__device__ __forceinline__ unsigned lds_lo32(const void* p) {
    return (unsigned)(unsigned long long)p;
}

// ---------------------------------------------------------------------------
// TDM: 2D tile load Global -> LDS.  D# per ISA 8.3/8.4 (data_size = 2 bytes).
// tensor = [td1 rows x td0 cols], row stride = stride0 elems, tile = tile1 x tile0.
// ---------------------------------------------------------------------------
__device__ __forceinline__ void tdm_load_2d(unsigned lds_addr, const void* gptr,
                                            unsigned td0, unsigned td1, unsigned stride0,
                                            unsigned tile0, unsigned tile1) {
    unsigned long long ga = (unsigned long long)gptr;
    v4u g0;
    g0[0] = 1u;                                                // count=1, user D#
    g0[1] = lds_addr;                                          // D#.lds_addr
    g0[2] = (unsigned)ga;                                      // global_addr[31:0]
    g0[3] = (unsigned)((ga >> 32) & 0x01FFFFFFu) | (2u << 30); // addr[56:32] | type=2
    v8i g1;
    g1[0] = 1 << 16;                                           // data_size = 2B
    g1[1] = (int)((td0 & 0xFFFFu) << 16);                      // tensor_dim0[15:0]
    g1[2] = (int)((td0 >> 16) | ((td1 & 0xFFFFu) << 16));      // dim0 hi | dim1 lo
    g1[3] = (int)((td1 >> 16) | (tile0 << 16));                // dim1 hi | tile_dim0
    g1[4] = (int)(tile1 & 0xFFFFu);                            // tile_dim1 (tile_dim2=0)
    g1[5] = (int)stride0;                                      // tensor_dim0_stride lo32
    g1[6] = 0;                                                 // stride0 hi / stride1 lo
    g1[7] = 0;
    v4i z4 = {0, 0, 0, 0};
#if defined(__clang_major__) && (__clang_major__ >= 23)
    v8i z8 = {0, 0, 0, 0, 0, 0, 0, 0};
    __builtin_amdgcn_tensor_load_to_lds(g0, g1, z4, z4, z8, 0);
#else
    __builtin_amdgcn_tensor_load_to_lds(g0, g1, z4, z4, 0);
#endif
}

// ---------------------------------------------------------------------------
// Kernel 1: weights -> bf16, zero the loss accumulator
// ---------------------------------------------------------------------------
__global__ void convert_weights(const float* __restrict__ wqkv,
                                const float* __restrict__ wout,
                                unsigned short* __restrict__ wq_bf,
                                unsigned short* __restrict__ wo_bf,
                                float* __restrict__ loss_acc) {
    int idx = blockIdx.x * blockDim.x + threadIdx.x;
    if (idx == 0) *loss_acc = 0.0f;
    const int n1 = D_ * NQKV_, n2 = D_ * D_;
    for (int i = idx; i < n1 + n2; i += gridDim.x * blockDim.x) {
        if (i < n1) wq_bf[i] = f32_to_bf16(wqkv[i]);
        else        wo_bf[i - n1] = f32_to_bf16(wout[i - n1]);
    }
}

// ---------------------------------------------------------------------------
// Kernel 2: LayerNorm + FiLM, one wave32 per row, output bf16
// ---------------------------------------------------------------------------
__global__ __launch_bounds__(256) void ln_film(const float* __restrict__ x,
                                               const float* __restrict__ gamma,
                                               const float* __restrict__ beta,
                                               const float* __restrict__ ln_w,
                                               const float* __restrict__ ln_b,
                                               unsigned short* __restrict__ xn) {
    int lane = threadIdx.x & 31, wid = threadIdx.x >> 5;
    int row = blockIdx.x * 8 + wid;                 // 8192 rows
    const float* xr = x + (long)row * D_;
    float s = 0.f, ss = 0.f;
#pragma unroll
    for (int e = 0; e < 16; ++e) {
        float v = xr[lane + 32 * e];
        s += v; ss += v * v;
    }
#pragma unroll
    for (int m = 1; m < 32; m <<= 1) { s += __shfl_xor(s, m); ss += __shfl_xor(ss, m); }
    float mu = s * (1.0f / D_);
    float var = ss * (1.0f / D_) - mu * mu;
    float rinv = rsqrtf(var + 1e-5f);
    int b = row >> 11;
#pragma unroll
    for (int e = 0; e < 16; ++e) {
        int c = lane + 32 * e;
        float v = (xr[c] - mu) * rinv * ln_w[c] + ln_b[c];
        v = gamma[b * D_ + c] * v + beta[b * D_ + c];
        xn[(long)row * D_ + c] = f32_to_bf16(v);
    }
}

// ---------------------------------------------------------------------------
// Kernel 3: QKV GEMM  [8192x512] x [512x1536], one wave per 16x16 tile.
// Epilogue scatters into Q[bh][n][dh] (pre-scaled), Kt[bh][dh][n], V[bh][n][dh].
// ---------------------------------------------------------------------------
__global__ __launch_bounds__(128) void qkv_gemm(const unsigned short* __restrict__ xn,
                                                const unsigned short* __restrict__ wq,
                                                unsigned short* __restrict__ Q,
                                                unsigned short* __restrict__ Kt,
                                                unsigned short* __restrict__ V) {
    int lane = threadIdx.x & 31, wid = threadIdx.x >> 5;
    int tile = blockIdx.x * 4 + wid;                // 512*96 tiles
    int mt = tile / 96, nt = tile % 96;
    int grp = lane >> 4, ln = lane & 15;
    const unsigned short* arow = xn + (long)(mt * 16 + ln) * D_;
    v8f acc = {0, 0, 0, 0, 0, 0, 0, 0};
#pragma unroll 4
    for (int k0 = 0; k0 < D_; k0 += 32) {
        v16bf a = load_a16(arow, k0, grp);
        v16bf b = load_b16(wq + (long)k0 * NQKV_ + nt * 16 + ln, NQKV_, grp);
        acc = wmma_bf16(a, b, acc);
    }
    const float qscale = 0.08838834764831845f;      // dh^-0.5
#pragma unroll
    for (int i = 0; i < 8; ++i) {
        int r = mt * 16 + i + 8 * grp;
        int c = nt * 16 + ln;
        int sel = c / D_, dcol = c % D_;
        int head = dcol >> 7, hcol = dcol & (DH_ - 1);
        int bb = r >> 11, ii = r & (N_ - 1);
        int bh = bb * H_ + head;
        float val = acc[i];
        if (sel == 0)      Q[((long)bh * N_ + ii) * DH_ + hcol]  = f32_to_bf16(val * qscale);
        else if (sel == 1) Kt[((long)bh * DH_ + hcol) * N_ + ii] = f32_to_bf16(val);
        else               V[((long)bh * N_ + ii) * DH_ + hcol]  = f32_to_bf16(val);
    }
}

// ---------------------------------------------------------------------------
// Kernel 4: flash attention + ALiBi + online entropy.
// Block = 8 waves = 128 query rows of one (b,h).  K/V j-tiles (32 wide) are
// DMA'd into LDS by the Tensor Data Mover (double-buffered, TENSORcnt), and
// all 8 waves consume them via ds_load fragments -> 8x less L2 traffic.
// ---------------------------------------------------------------------------
__global__ __launch_bounds__(256) void attn(const unsigned short* __restrict__ Q,
                                            const unsigned short* __restrict__ Kt,
                                            const unsigned short* __restrict__ V,
                                            const float* __restrict__ alibi,
                                            unsigned short* __restrict__ ctx,
                                            float* __restrict__ loss_acc) {
    __shared__ __align__(32) unsigned short kbuf[2][DH_ * 32];  // [dh][j] 8KB each
    __shared__ __align__(32) unsigned short vbuf[2][32 * DH_];  // [j][dh] 8KB each
    __shared__ __align__(32) unsigned short lp[8][16 * 32];     // per-wave P staging
    int lane = threadIdx.x & 31, wid = threadIdx.x >> 5;
    int bh = blockIdx.x >> 4, qblk = blockIdx.x & 15;           // 16 q-blocks per (b,h)
    int qt = qblk * 8 + wid;                                    // 16-row tile index
    int b = bh >> 2, head = bh & 3;
    int grp = lane >> 4, ln = lane & 15;

    const unsigned short* KtB = Kt + (size_t)bh * DH_ * N_;
    const unsigned short* VB  = V  + (size_t)bh * N_ * DH_;

    v16bf aq[4];
#pragma unroll
    for (int ks = 0; ks < 4; ++ks)
        aq[ks] = load_a16(Q + ((size_t)bh * N_ + qt * 16 + ln) * DH_, ks * 32, grp);

    v8f o[8];
    float m[8], l[8], te[8];
#pragma unroll
    for (int c = 0; c < 8; ++c) o[c] = (v8f){0, 0, 0, 0, 0, 0, 0, 0};
#pragma unroll
    for (int i = 0; i < 8; ++i) { m[i] = -1e30f; l[i] = 0.f; te[i] = 0.f; }

    // prologue: TDM loads of tile 0 (Kt: 128x32 rows of 32 j; V: 32x128)
    if (wid == 0) {
        tdm_load_2d(lds_lo32(&kbuf[0][0]), KtB, N_, DH_, N_, 32, DH_);
        tdm_load_2d(lds_lo32(&vbuf[0][0]), VB,  DH_, N_, DH_, DH_, 32);
    }

    const int NT = N_ / 32;                          // 64 j-tiles
    for (int it = 0; it < NT; ++it) {
        int buf = it & 1;
        int jt = it * 32;
        if (wid == 0) __builtin_amdgcn_s_wait_tensorcnt(0);   // tile `it` in LDS
        __syncthreads();                                      // publish to all waves
        if (wid == 0 && it + 1 < NT) {                        // prefetch next tile,
            int nb = buf ^ 1, njt = jt + 32;                  // overlaps with compute
            tdm_load_2d(lds_lo32(&kbuf[nb][0]), KtB + njt,              N_, DH_, N_, 32, DH_);
            tdm_load_2d(lds_lo32(&vbuf[nb][0]), VB + (size_t)njt * DH_, DH_, N_, DH_, DH_, 32);
        }

        // scores: S[16 x 32] = Q (16x128) x K-tile, two 16-wide N sub-tiles
        v8f s0 = {0, 0, 0, 0, 0, 0, 0, 0}, s1 = {0, 0, 0, 0, 0, 0, 0, 0};
#pragma unroll
        for (int ks = 0; ks < 4; ++ks) {
            const unsigned short* kp = &kbuf[buf][(ks * 32) * 32 + ln];
            s0 = wmma_bf16(aq[ks], load_b16(kp, 32, grp), s0);
            s1 = wmma_bf16(aq[ks], load_b16(kp + 16, 32, grp), s1);
        }
        // + ALiBi (broadcast over heads): C-layout elem i -> row i+8*grp, col ln
#pragma unroll
        for (int i = 0; i < 8; ++i) {
            const float* ab = alibi + (((long)b * N_ + qt * 16 + i + 8 * grp) * N_ + jt + ln);
            s0[i] += ab[0];
            s1[i] += ab[16];
        }
        // online softmax: reduce over j (lanes 0..15 of each half)
        float alpha_[8];
#pragma unroll
        for (int i = 0; i < 8; ++i) {
            float v = fmaxf(s0[i], s1[i]);
#pragma unroll
            for (int msk = 1; msk < 16; msk <<= 1) v = fmaxf(v, __shfl_xor(v, msk));
            float mnew = fmaxf(m[i], v);
            float al = __expf(m[i] - mnew);
            float p0 = __expf(s0[i] - mnew);
            float p1 = __expf(s1[i] - mnew);
            float psum = p0 + p1;
            float tl = p0 * (s0[i] - mnew) + p1 * (s1[i] - mnew);
#pragma unroll
            for (int msk = 1; msk < 16; msk <<= 1) {
                psum += __shfl_xor(psum, msk);
                tl += __shfl_xor(tl, msk);
            }
            te[i] = al * (te[i] + (m[i] - mnew) * l[i]) + tl;
            l[i] = al * l[i] + psum;
            m[i] = mnew;
            alpha_[i] = al;
            int prow = i + 8 * grp;
            lp[wid][prow * 32 + ln] = f32_to_bf16(p0);        // C-layout -> LDS
            lp[wid][prow * 32 + 16 + ln] = f32_to_bf16(p1);
        }
#pragma unroll
        for (int c = 0; c < 8; ++c)
#pragma unroll
            for (int i = 0; i < 8; ++i) o[c][i] *= alpha_[i];
        // per-wave LDS RAW fence (DS ops in-order per wave; keep compiler honest)
        asm volatile("s_wait_dscnt 0" ::: "memory");
        v16bf pa = load_a16(&lp[wid][ln * 32], 0, grp);       // A-layout P (16x32)
        // O += P (16x32) x V-tile (32x128): B lane = dh col, elem = j_local
        const unsigned short* vp = &vbuf[buf][ln];
#pragma unroll
        for (int c = 0; c < 8; ++c)
            o[c] = wmma_bf16(pa, load_b16(vp + c * 16, DH_, grp), o[c]);
        __syncthreads();                 // reads of `buf` done before TDM reuses it
    }

    // epilogue: normalize, write ctx (bf16), accumulate entropy
#pragma unroll
    for (int i = 0; i < 8; ++i) {
        float inv = 1.0f / l[i];
        long row = (long)b * N_ + qt * 16 + i + 8 * grp;
#pragma unroll
        for (int c = 0; c < 8; ++c)
            ctx[row * D_ + head * DH_ + c * 16 + ln] = f32_to_bf16(o[c][i] * inv);
    }
    if (ln == 0) {                                  // stats replicated per half-wave
        float hs = 0.f;
#pragma unroll
        for (int i = 0; i < 8; ++i) hs += logf(l[i]) - te[i] / l[i];
        atomicAdd(loss_acc, hs);
    }
}

// ---------------------------------------------------------------------------
// Kernel 5: out projection [8192x512] x [512x512] -> fp32 d_out
// ---------------------------------------------------------------------------
__global__ __launch_bounds__(128) void out_gemm(const unsigned short* __restrict__ ctx,
                                                const unsigned short* __restrict__ wo,
                                                float* __restrict__ out) {
    int lane = threadIdx.x & 31, wid = threadIdx.x >> 5;
    int tile = blockIdx.x * 4 + wid;                // 512*32 tiles
    int mt = tile / 32, nt = tile % 32;
    int grp = lane >> 4, ln = lane & 15;
    const unsigned short* arow = ctx + (long)(mt * 16 + ln) * D_;
    v8f acc = {0, 0, 0, 0, 0, 0, 0, 0};
#pragma unroll 4
    for (int k0 = 0; k0 < D_; k0 += 32) {
        v16bf a = load_a16(arow, k0, grp);
        v16bf b = load_b16(wo + (long)k0 * D_ + nt * 16 + ln, D_, grp);
        acc = wmma_bf16(a, b, acc);
    }
#pragma unroll
    for (int i = 0; i < 8; ++i) {
        long r = mt * 16 + i + 8 * grp;
        out[r * D_ + nt * 16 + ln] = acc[i];
    }
}

__global__ void finalize_loss(const float* __restrict__ loss_acc, float* __restrict__ dst) {
    if (threadIdx.x == 0 && blockIdx.x == 0)
        *dst = (*loss_acc) * (0.1f / (float)(B_ * H_ * N_));
}

// ---------------------------------------------------------------------------
extern "C" void kernel_launch(void* const* d_in, const int* in_sizes, int n_in,
                              void* d_out, int out_size, void* d_ws, size_t ws_size,
                              hipStream_t stream) {
    (void)in_sizes; (void)n_in; (void)out_size; (void)ws_size;
    const float* x     = (const float*)d_in[0];
    const float* alibi = (const float*)d_in[1];
    const float* gamma = (const float*)d_in[2];
    const float* beta  = (const float*)d_in[3];
    const float* ln_w  = (const float*)d_in[4];
    const float* ln_b  = (const float*)d_in[5];
    const float* wqkv  = (const float*)d_in[6];
    const float* wout  = (const float*)d_in[7];
    float* out = (float*)d_out;

    char* ws = (char*)d_ws;
    size_t off = 0;
    unsigned short* xn = (unsigned short*)(ws + off); off += (size_t)ROWS_ * D_ * 2;
    unsigned short* wq = (unsigned short*)(ws + off); off += (size_t)D_ * NQKV_ * 2;
    unsigned short* wo = (unsigned short*)(ws + off); off += (size_t)D_ * D_ * 2;
    unsigned short* Q  = (unsigned short*)(ws + off); off += (size_t)B_ * H_ * N_ * DH_ * 2;
    unsigned short* Kt = (unsigned short*)(ws + off); off += (size_t)B_ * H_ * N_ * DH_ * 2;
    unsigned short* V  = (unsigned short*)(ws + off); off += (size_t)B_ * H_ * N_ * DH_ * 2;
    unsigned short* ctx = (unsigned short*)(ws + off); off += (size_t)ROWS_ * D_ * 2;
    float* loss_acc = (float*)(ws + off);

    convert_weights<<<256, 256, 0, stream>>>(wqkv, wout, wq, wo, loss_acc);
    ln_film<<<ROWS_ / 8, 256, 0, stream>>>(x, gamma, beta, ln_w, ln_b, xn);
    qkv_gemm<<<(ROWS_ / 16) * (NQKV_ / 16) / 4, 128, 0, stream>>>(xn, wq, Q, Kt, V);
    attn<<<B_ * H_ * 16, 256, 0, stream>>>(Q, Kt, V, alibi, ctx, loss_acc);
    out_gemm<<<(ROWS_ / 16) * (D_ / 16) / 4, 128, 0, stream>>>(ctx, wo, out);
    finalize_loss<<<1, 32, 0, stream>>>(loss_acc, out + (size_t)ROWS_ * D_);
}
